// ElectronGNNLayer_22600117911703
// MI455X (gfx1250) — matrix operand order repaired and checked by reading, not compiled, not run on hardware
//
#include <hip/hip_runtime.h>
#include <hip/hip_bf16.h>

typedef __attribute__((ext_vector_type(16))) _Float16 v16h;
typedef __attribute__((ext_vector_type(8)))  float    v8f;

#define NW 8          // waves per block (wave32)
#define NT 256        // threads per block
#define N_ELEC 32
#define N_NUC  8
#define EMB    256
#define TPS    32
#define FDIM   352    // EMB + 3*TPS

// ---- pre-swizzled f16 weight fragments in d_ws ----
// One fragment = one WMMA B tile (32x16) = 32 lanes x 16 f16 = 1 KB.
// Element e of lane l of fragment (W,k0,n0):  W[k0 + (l>>4)*16 + e][n0 + (l&15)]
#define FR_UW1 0      //  6 frags: [t*2 + nc]
#define FR_UW2 6      //  6
#define FR_WW1 12     //  6
#define FR_WW2 18     //  6
#define FR_HW2 24     //  6: [t*2 + nc]
#define FR_HW1 30     // 48: [t*16 + ks*2 + nc]
#define FR_GW1 78     // 176: [ks*16 + nc]   (K=352 -> 11 ktiles, N=256 -> 16 ntiles)
#define FR_GW2 254    // 128: [ks*16 + nc]
#define TOTAL_FRAGS 382

static __device__ __forceinline__ float fast_tanh(float x) {
#if defined(__has_builtin) && __has_builtin(__builtin_amdgcn_tanhf)
  return __builtin_amdgcn_tanhf(x);   // v_tanh_f32 (gfx1250 trans op)
#else
  return tanhf(x);
#endif
}

static __device__ __forceinline__ v8f wmma16(v16h a, v16h b, v8f c) {
  // D = A(16x32,f16) * B(32x16,f16) + C(16x16,f32)
  return __builtin_amdgcn_wmma_f32_16x16x32_f16(false, a, false, b, (short)0, c,
                                                false, false);
}

// B fragment: one contiguous 32B vector per lane, fully coalesced.
static __device__ __forceinline__ v16h load_frag(const _Float16* __restrict__ wf,
                                                 int frag, int lane) {
  return *((const v16h*)(wf + ((size_t)frag << 9)) + lane);
}

// A tile (16x32 f16) from row-major f16 src with row stride ld.
static __device__ __forceinline__ v16h load_a_h(const _Float16* src, int ld, int lane) {
  int m = lane & 15, hfl = lane >> 4;
  v16h a;
#pragma unroll
  for (int v = 0; v < 8; ++v) {
    int k = (hfl << 4) + (v << 1);
    a[2*v]   = src[m*ld + k];
    a[2*v+1] = src[m*ld + k + 1];
  }
  return a;
}

// A tile from row-major f32 src (convert to f16); rows clamped to maxm (8-row nuclei).
static __device__ __forceinline__ v16h load_a_f(const float* src, int ld, int lane, int maxm) {
  int m = lane & 15; if (m > maxm) m = maxm;
  int hfl = lane >> 4;
  v16h a;
#pragma unroll
  for (int v = 0; v < 8; ++v) {
    int k = (hfl << 4) + (v << 1);
    a[2*v]   = (_Float16)src[m*ld + k];
    a[2*v+1] = (_Float16)src[m*ld + k + 1];
  }
  return a;
}

// ---------------- prep kernel: f32 weights -> f16 B-fragments in ws ----------------
__global__ __launch_bounds__(NT) void prep_weights_kernel(
    const float* __restrict__ u_w1, const float* __restrict__ u_w2,
    const float* __restrict__ w_w1, const float* __restrict__ w_w2,
    const float* __restrict__ h_w1, const float* __restrict__ h_w2,
    const float* __restrict__ g_w1, const float* __restrict__ g_w2,
    _Float16* __restrict__ wf)
{
  int tid = blockIdx.x * NT + threadIdx.x;
  const int total = TOTAL_FRAGS * 512;
  if (tid >= total) return;
  int frag = tid >> 9;
  int r    = tid & 511;
  int lane = r >> 4;
  int e    = r & 15;

  const float* W; int ldn, k0, n0;
  if (frag < 24) {                       // u_w1,u_w2,w_w1,w_w2 (6 frags each)
    int m = frag / 6, rem = frag % 6;
    int t = rem >> 1, nc = rem & 1;
    const float* base = (m == 0) ? u_w1 : (m == 1) ? u_w2 : (m == 2) ? w_w1 : w_w2;
    W = base + t * TPS * TPS; ldn = TPS; k0 = 0; n0 = nc * 16;
  } else if (frag < 30) {                // h_w2
    int rem = frag - FR_HW2; int t = rem >> 1, nc = rem & 1;
    W = h_w2 + t * TPS * TPS; ldn = TPS; k0 = 0; n0 = nc * 16;
  } else if (frag < 78) {                // h_w1
    int rem = frag - FR_HW1; int t = rem >> 4; int rr = rem & 15;
    int ks = rr >> 1, nc = rr & 1;
    W = h_w1 + t * EMB * TPS; ldn = TPS; k0 = ks * TPS; n0 = nc * 16;
  } else if (frag < 254) {               // g_w1 (352x256)
    int rem = frag - FR_GW1; int ks = rem >> 4, nc = rem & 15;
    W = g_w1; ldn = EMB; k0 = ks * TPS; n0 = nc * 16;
  } else {                               // g_w2 (256x256)
    int rem = frag - FR_GW2; int ks = rem >> 4, nc = rem & 15;
    W = g_w2; ldn = EMB; k0 = ks * TPS; n0 = nc * 16;
  }
  int hf = lane >> 4, nn = lane & 15;
  wf[tid] = (_Float16)W[(k0 + hf * 16 + e) * ldn + n0 + nn];
}

// ---------------- fused per-sample kernel ----------------
// Static LDS plan (62 KB):
struct SMem {
  float    z[3][N_ELEC][TPS];     // 12 KB  scatter accumulators (LDS atomics)
  float    hx[N_ELEC][TPS];       //  4 KB  node-MLP output for current edge type
  _Float16 stage[NW][16][TPS];    //  8 KB  per-wave C->A transpose staging
  _Float16 f[N_ELEC][FDIM];       // 22 KB  concatenated features (f16)
  _Float16 hidden[N_ELEC][EMB];   // 16 KB  g-MLP hidden (f16)
};

__global__ __launch_bounds__(NT) void egnn_fused_kernel(
    const float* __restrict__ electrons, const float* __restrict__ nuclei,
    const float* __restrict__ feat_same, const float* __restrict__ feat_anti,
    const float* __restrict__ feat_ne,
    const float* __restrict__ u_b1, const float* __restrict__ u_b2,
    const float* __restrict__ w_b1, const float* __restrict__ w_b2,
    const float* __restrict__ h_b1, const float* __restrict__ h_b2,
    const float* __restrict__ g_b1, const float* __restrict__ g_b2,
    const _Float16* __restrict__ wf,
    const int* __restrict__ ss, const int* __restrict__ rs,
    const int* __restrict__ sa, const int* __restrict__ ra,
    const int* __restrict__ sn, const int* __restrict__ rn,
    float* __restrict__ out)
{
  __shared__ SMem sm;
  const int b    = blockIdx.x;
  const int lane = threadIdx.x & 31;
  const int wave = threadIdx.x >> 5;
  const int n    = lane & 15;
  const int hfl  = lane >> 4;

  const float* eb = electrons + (size_t)b * N_ELEC * EMB;
  const float* nb = nuclei    + (size_t)b * N_NUC  * EMB;

  const float* featp[3] = { feat_same + (size_t)b * 480 * TPS,
                            feat_anti + (size_t)b * 512 * TPS,
                            feat_ne   + (size_t)b * 256 * TPS };
  const int  Ecnt[3] = {480, 512, 256};
  const int* Sp[3]   = {ss, sa, sn};
  const int* Rp[3]   = {rs, ra, rn};

  for (int i = threadIdx.x; i < 3 * N_ELEC * TPS; i += NT)
    ((float*)sm.z)[i] = 0.0f;

#pragma unroll
  for (int t = 0; t < 3; ++t) {
    __syncthreads();   // z zeroed / previous type's hx reads finished

    // ---------- node MLP: hx = mlp(sender_nodes, h[t]) ----------
    const int rowTiles = (t < 2) ? 2 : 1;
    if (wave < rowTiles) {
      const float* nodes = (t < 2) ? eb : nb;
      const int    maxm  = (t < 2) ? 15 : 7;
      const float* hb1 = h_b1 + t * TPS;
      const float* hb2 = h_b2 + t * TPS;
      const int rt = wave;
      _Float16* st = &sm.stage[wave][0][0];
#pragma unroll
      for (int nc = 0; nc < 2; ++nc) {          // layer 1: (16x256)@(256x32)
        v8f acc = {};
#pragma unroll
        for (int ks = 0; ks < EMB / TPS; ++ks) {
          v16h a  = load_a_f(nodes + rt*16*EMB + ks*TPS, EMB, lane, maxm);
          v16h bm = load_frag(wf, FR_HW1 + t*16 + ks*2 + nc, lane);
          acc = wmma16(a, bm, acc);
        }
        float bias = hb1[nc*16 + n];
#pragma unroll
        for (int i = 0; i < 8; ++i)
          st[(i + 8*hfl)*TPS + nc*16 + n] = (_Float16)fast_tanh(acc[i] + bias);
      }
      v16h a2 = load_a_h(st, TPS, lane);        // layer 2: (16x32)@(32x32)
#pragma unroll
      for (int nc = 0; nc < 2; ++nc) {
        v16h bm = load_frag(wf, FR_HW2 + t*2 + nc, lane);
        v8f acc = {};
        acc = wmma16(a2, bm, acc);
        float bias = hb2[nc*16 + n];
#pragma unroll
        for (int i = 0; i < 8; ++i)
          sm.hx[rt*16 + i + 8*hfl][nc*16 + n] = acc[i] + bias;
      }
    }
    __syncthreads();   // hx ready for all waves

    // ---------- edge MLPs + scatter into z ----------
    const float* ub1 = u_b1 + t*TPS; const float* ub2 = u_b2 + t*TPS;
    const float* wb1 = w_b1 + t*TPS; const float* wb2 = w_b2 + t*TPS;

    v16h Bu1[2], Bu2[2], Bw1[2], Bw2[2];
    float bu1[2], bu2[2], bw1v[2], bw2v[2];
#pragma unroll
    for (int nc = 0; nc < 2; ++nc) {
      Bu1[nc] = load_frag(wf, FR_UW1 + t*2 + nc, lane);
      Bu2[nc] = load_frag(wf, FR_UW2 + t*2 + nc, lane);
      Bw1[nc] = load_frag(wf, FR_WW1 + t*2 + nc, lane);
      Bw2[nc] = load_frag(wf, FR_WW2 + t*2 + nc, lane);
      bu1[nc]  = ub1[nc*16 + n]; bu2[nc]  = ub2[nc*16 + n];
      bw1v[nc] = wb1[nc*16 + n]; bw2v[nc] = wb2[nc*16 + n];
    }

    const int  T  = Ecnt[t] / 16;
    const int* St = Sp[t];
    const int* Rt = Rp[t];
    _Float16*  st = &sm.stage[wave][0][0];

    for (int tile = wave; tile < T; tile += NW) {
      const float* ft = featp[t] + tile*16*TPS;
      if (tile + NW < T)   // prefetch next tile (global_prefetch_b8)
        __builtin_prefetch(featp[t] + (tile + NW)*16*TPS, 0, 0);
      v16h a0 = load_a_f(ft, TPS, lane, 15);
      // u layer 1: tanh(feat@u_w1 + u_b1)
#pragma unroll
      for (int nc = 0; nc < 2; ++nc) {
        v8f c = {};
        c = wmma16(a0, Bu1[nc], c);
#pragma unroll
        for (int i = 0; i < 8; ++i)
          st[(i + 8*hfl)*TPS + nc*16 + n] = (_Float16)fast_tanh(c[i] + bu1[nc]);
      }
      v16h a1 = load_a_h(st, TPS, lane);
      // u layer 2 + residual -> e = feat + mlp(feat)
#pragma unroll
      for (int nc = 0; nc < 2; ++nc) {
        v8f c = {};
        c = wmma16(a1, Bu2[nc], c);
#pragma unroll
        for (int i = 0; i < 8; ++i) {
          float e = c[i] + bu2[nc] + ft[(i + 8*hfl)*TPS + nc*16 + n];
          st[(i + 8*hfl)*TPS + nc*16 + n] = (_Float16)e;
        }
      }
      v16h a2 = load_a_h(st, TPS, lane);
      // w layer 1: tanh(e@w_w1 + w_b1)
#pragma unroll
      for (int nc = 0; nc < 2; ++nc) {
        v8f c = {};
        c = wmma16(a2, Bw1[nc], c);
#pragma unroll
        for (int i = 0; i < 8; ++i)
          st[(i + 8*hfl)*TPS + nc*16 + n] = (_Float16)fast_tanh(c[i] + bw1v[nc]);
      }
      v16h a3 = load_a_h(st, TPS, lane);
      int sidx[8], ridx[8];
#pragma unroll
      for (int i = 0; i < 8; ++i) {
        int erow = tile*16 + i + 8*hfl;
        sidx[i] = St[erow];
        ridx[i] = Rt[erow];
      }
      // w layer 2 -> we ; msg = we * hx[sender] ; z[receiver] += msg
#pragma unroll
      for (int nc = 0; nc < 2; ++nc) {
        v8f c = {};
        c = wmma16(a3, Bw2[nc], c);
#pragma unroll
        for (int i = 0; i < 8; ++i) {
          float we  = c[i] + bw2v[nc];
          float val = we * sm.hx[sidx[i]][nc*16 + n];
          atomicAdd(&sm.z[t][ridx[i]][nc*16 + n], val);
        }
      }
    }
  }
  __syncthreads();

  // ---------- f = [elec | z_ne/8 | z_same/15 | z_anti/16] (f16) ----------
  for (int idx = threadIdx.x; idx < N_ELEC * FDIM; idx += NT) {
    int row = idx / FDIM, col = idx - row * FDIM;
    float v;
    if      (col < EMB)           v = eb[row*EMB + col];
    else if (col < EMB + TPS)     v = sm.z[2][row][col - EMB]          * 0.125f;
    else if (col < EMB + 2*TPS)   v = sm.z[0][row][col - EMB - TPS]    * (1.0f/15.0f);
    else                          v = sm.z[1][row][col - EMB - 2*TPS]  * 0.0625f;
    sm.f[row][col] = (_Float16)v;
  }
  __syncthreads();

  // ---------- g layer 1: hidden = tanh(f @ g_w1 + g_b1) ----------
  for (int task = wave; task < 32; task += NW) {
    int rt = task >> 4, nc = task & 15;
    v8f acc = {};
#pragma unroll
    for (int ks = 0; ks < FDIM / TPS; ++ks) {
      v16h a  = load_a_h(&sm.f[rt*16][ks*TPS], FDIM, lane);
      v16h bm = load_frag(wf, FR_GW1 + ks*16 + nc, lane);
      acc = wmma16(a, bm, acc);
    }
    float bias = g_b1[nc*16 + n];
#pragma unroll
    for (int i = 0; i < 8; ++i)
      sm.hidden[rt*16 + i + 8*hfl][nc*16 + n] = (_Float16)fast_tanh(acc[i] + bias);
  }
  __syncthreads();

  // ---------- g layer 2 + residual -> out ----------
  float* ob = out + (size_t)b * N_ELEC * EMB;
  for (int task = wave; task < 32; task += NW) {
    int rt = task >> 4, nc = task & 15;
    v8f acc = {};
#pragma unroll
    for (int ks = 0; ks < EMB / TPS; ++ks) {
      v16h a  = load_a_h(&sm.hidden[rt*16][ks*TPS], EMB, lane);
      v16h bm = load_frag(wf, FR_GW2 + ks*16 + nc, lane);
      acc = wmma16(a, bm, acc);
    }
    float bias = g_b2[nc*16 + n];
#pragma unroll
    for (int i = 0; i < 8; ++i) {
      int row = rt*16 + i + 8*hfl, col = nc*16 + n;
      ob[row*EMB + col] = eb[row*EMB + col] + acc[i] + bias;
    }
  }
}

extern "C" void kernel_launch(void* const* d_in, const int* in_sizes, int n_in,
                              void* d_out, int out_size, void* d_ws, size_t ws_size,
                              hipStream_t stream) {
  const float* electrons = (const float*)d_in[0];
  const float* nuclei    = (const float*)d_in[1];
  const float* feat_same = (const float*)d_in[2];
  const float* feat_anti = (const float*)d_in[3];
  const float* feat_ne   = (const float*)d_in[4];
  const float* u_w1 = (const float*)d_in[5];
  const float* u_b1 = (const float*)d_in[6];
  const float* u_w2 = (const float*)d_in[7];
  const float* u_b2 = (const float*)d_in[8];
  const float* w_w1 = (const float*)d_in[9];
  const float* w_b1 = (const float*)d_in[10];
  const float* w_w2 = (const float*)d_in[11];
  const float* w_b2 = (const float*)d_in[12];
  const float* h_w1 = (const float*)d_in[13];
  const float* h_b1 = (const float*)d_in[14];
  const float* h_w2 = (const float*)d_in[15];
  const float* h_b2 = (const float*)d_in[16];
  const float* g_w1 = (const float*)d_in[17];
  const float* g_b1 = (const float*)d_in[18];
  const float* g_w2 = (const float*)d_in[19];
  const float* g_b2 = (const float*)d_in[20];
  const int* ss = (const int*)d_in[21];
  const int* rs = (const int*)d_in[22];
  const int* sa = (const int*)d_in[23];
  const int* ra = (const int*)d_in[24];
  const int* sn = (const int*)d_in[25];
  const int* rn = (const int*)d_in[26];
  float* out = (float*)d_out;
  _Float16* wf = (_Float16*)d_ws;   // 382 KB of pre-swizzled f16 fragments

  const int Bn = in_sizes[0] / (N_ELEC * EMB);   // batch size (2048)

  const int prep_total = TOTAL_FRAGS * 512;
  prep_weights_kernel<<<(prep_total + NT - 1) / NT, NT, 0, stream>>>(
      u_w1, u_w2, w_w1, w_w2, h_w1, h_w2, g_w1, g_w2, wf);

  egnn_fused_kernel<<<Bn, NT, 0, stream>>>(
      electrons, nuclei, feat_same, feat_anti, feat_ne,
      u_b1, u_b2, w_b1, w_b2, h_b1, h_b2, g_b1, g_b2, wf,
      ss, rs, sa, ra, sn, rn, out);
}
